// SceneEncoder_455266533848
// MI455X (gfx1250) — matrix operand order repaired and checked by reading, not compiled, not run on hardware
//
#include <hip/hip_runtime.h>

typedef __attribute__((ext_vector_type(16))) __bf16 v16bf;
typedef __attribute__((ext_vector_type(8)))  __bf16 v8bf;
typedef __attribute__((ext_vector_type(8)))  float  v8f;

#define NBATCH 8
#define NG     512      // B*T graphs
#define NN     128      // nodes per graph
#define NEDGE  512      // edges per graph
#define HID    256
#define NH     8
#define HD     32
#define CCAT   11
#define SSEQ   65
#define MROWS  520      // 8*65
#define MPAD   528
#define DFF    2048
#define CLIPV  1.0e6f

// ---------------------------------------------------------------- helpers

__device__ __forceinline__ float sanit(float v) {
  if (v != v) return 0.f;                       // nan -> 0
  return fminf(fmaxf(v, -CLIPV), CLIPV);        // +-inf clipped too
}

// Load A fragments (f32 in LDS -> bf16) for this wave's fixed 16-row M-tile.
// A fragment layout: lane half=lane>>4, m=lane&15;
//   elem e -> K = k0 + 16*half + e
__device__ __forceinline__ void load_afrags(const float* A, int lda, int rbase,
                                            v16bf* frag, int ksteps) {
  const int lane = threadIdx.x & 31;
  const int half = lane >> 4;
  const int m = lane & 15;
  for (int s = 0; s < ksteps; ++s) {
    const float* pa = A + (rbase + m) * lda + s * 32 + 16 * half;
    v16bf a;
#pragma unroll
    for (int e = 0; e < 8; ++e) {
      a[e]     = (__bf16)pa[e];
      a[e + 8] = (__bf16)pa[e + 8];
    }
    frag[s] = a;
  }
}

// C += A(preloaded frags) x B(bf16, [N][K] transposed in global).
// B fragment: col = lane&15 ; elem e -> K = k0 + 16*half + e (contiguous 32B)
__device__ __forceinline__ v8f wtile_pre(const v16bf* frag, int ksteps,
                                         const __bf16* Bt, int ldb,
                                         int cbase, int kB0, v8f acc) {
  const int lane = threadIdx.x & 31;
  const int half = lane >> 4;
  const int m = lane & 15;
  for (int s = 0; s < ksteps; ++s) {
    const __bf16* pb = Bt + (size_t)(cbase + m) * ldb + kB0 + s * 32 + 16 * half;
    v16bf b = *(const v16bf*)pb;
    acc = __builtin_amdgcn_wmma_f32_16x16x32_bf16(false, frag[s], false, b, (short)0, acc,
                                                  false, false);
  }
  return acc;
}

// Store a D tile (f32 acc) into LDS as bf16. D layout: VGPR r -> row r+8*half, col lane&15.
__device__ __forceinline__ void store_tile_bf16(__bf16* O, int ldo, int rbase, int cbase, v8f acc) {
  const int lane = threadIdx.x & 31;
  const int half = lane >> 4;
  const int m = lane & 15;
#pragma unroll
  for (int r = 0; r < 8; ++r)
    O[(rbase + r + 8 * half) * ldo + cbase + m] = (__bf16)acc[r];
}

// ---------------------------------------------------------------- weight prep: f32 [K][N] -> bf16 [N][K]
__global__ void k_w2bf_t(const float* W, unsigned short* Wt_, int K, int N) {
  int idx = blockIdx.x * 256 + threadIdx.x;
  if (idx >= K * N) return;
  int k = idx / N, n = idx % N;
  ((__bf16*)Wt_)[(size_t)n * K + k] = (__bf16)W[(size_t)k * N + n];
}

// ---------------------------------------------------------------- fused spatial GNN: 1 workgroup per graph
__global__ __launch_bounds__(256)
void k_spatial(const float* X, const int* EI, const int* ET, const float* GEO,
               const float* Win, const float* bin,
               const unsigned short* WqT_, const unsigned short* WkT_,
               const unsigned short* WvT_, const unsigned short* WoT_,
               const float* Wg, const float* Wc, const float* bc, float* Z) {
  __shared__ float   sh_h[NN * HID];          // 128 KB, f32 h (persistent)
  __shared__ __bf16  sh_q[NN * HID];          // 64 KB, q then v
  __shared__ float   sh_kagg[NN * HID / 2];   // 64 KB, bf16 k  OR  f32 agg half [128][128]
  __shared__ float   sh_e[NEDGE * NH];        // 16 KB, logits/ex ; later csum [11][256]
  __shared__ unsigned sh_m[NN * NH];          // 4 KB, encoded segment max
  __shared__ float   sh_d[NN * NH];           // 4 KB, denom ; later cnt[11]
  __shared__ float   sh_xin[NN * 16];         // 8 KB, sanitized x
  __shared__ float   sh_geo[NN * 3];          // sanitized geo

  const int g    = blockIdx.x;
  const int tid  = threadIdx.x;
  const int wave = tid >> 5;
  const int lane = tid & 31;
  const int half = lane >> 4;
  const int mlan = lane & 15;
  __bf16* sh_k = (__bf16*)sh_kagg;

  // ---- phase 0: load + sanitize inputs, h = relu(x @ W_in + b_in)
  for (int i = tid; i < NN * 16; i += 256) sh_xin[i] = sanit(X[(size_t)g * NN * 16 + i]);
  for (int i = tid; i < NN * 3; i += 256)  sh_geo[i] = sanit(GEO[(size_t)g * NN * 3 + i]);
  __syncthreads();
  for (int oi = tid; oi < NN * HID; oi += 256) {
    int node = oi >> 8, col = oi & 255;
    float s = bin[col];
    for (int j = 0; j < 16; ++j) s += sh_xin[node * 16 + j] * Win[j * HID + col];
    sh_h[oi] = fmaxf(s, 0.f);
  }
  __syncthreads();

  const int* srcp = EI + (size_t)g * 2 * NEDGE;
  const int* dstp = srcp + NEDGE;

  // ---- 3 spatial attention layers
  for (int l = 0; l < 3; ++l) {
    const __bf16* Wq = (const __bf16*)WqT_ + (size_t)l * HID * HID;
    const __bf16* Wk = (const __bf16*)WkT_ + (size_t)l * HID * HID;
    const __bf16* Wv = (const __bf16*)WvT_ + (size_t)l * HID * HID;
    const __bf16* Wo = (const __bf16*)WoT_ + (size_t)l * HID * HID;

    // hoist this wave's A fragments of h (reused for q, k and v GEMMs)
    v16bf hfr[8];
    load_afrags(sh_h, HID, wave * 16, hfr, 8);

    // q = h@Wq, k = h@Wk   (wave w owns M-tile row w, all 16 N-tiles)
    for (int i = 0; i < 16; ++i) {
      v8f a0 = {0.f, 0.f, 0.f, 0.f, 0.f, 0.f, 0.f, 0.f};
      a0 = wtile_pre(hfr, 8, Wq, HID, i * 16, 0, a0);
      store_tile_bf16(sh_q, HID, wave * 16, i * 16, a0);
      v8f a1 = {0.f, 0.f, 0.f, 0.f, 0.f, 0.f, 0.f, 0.f};
      a1 = wtile_pre(hfr, 8, Wk, HID, i * 16, 0, a1);
      store_tile_bf16(sh_k, HID, wave * 16, i * 16, a1);
    }
    __syncthreads();

    // stats init
    for (int i = tid; i < NN * NH; i += 256) { sh_m[i] = 0u; sh_d[i] = 0.f; }
    __syncthreads();

    // logits + segment max (encoded uint)
    const float* wg = Wg + l * 3 * NH;
    const float invs = 0.17677669529663687f;  // 1/sqrt(32)
    for (int it = tid; it < NEDGE * NH; it += 256) {
      int e = it >> 3, hd = it & 7;
      int s = srcp[e] & 127, d = dstp[e] & 127;
      float dot = 0.f;
      for (int j = 0; j < HD; ++j)
        dot += (float)sh_q[d * HID + hd * HD + j] * (float)sh_k[s * HID + hd * HD + j];
      dot *= invs;
      for (int x = 0; x < 3; ++x)
        dot += (sh_geo[s * 3 + x] - sh_geo[d * 3 + x]) * wg[x * NH + hd];
      sh_e[it] = dot;
      unsigned u = __float_as_uint(dot);
      u = (u & 0x80000000u) ? ~u : (u | 0x80000000u);
      atomicMax(&sh_m[d * NH + hd], u);
    }
    __syncthreads();

    // exp(logit - m) + segment sum
    for (int it = tid; it < NEDGE * NH; it += 256) {
      int e = it >> 3, hd = it & 7;
      int d = dstp[e] & 127;
      unsigned u = sh_m[d * NH + hd];
      float mv = 0.f;
      if (u != 0u) {
        unsigned w2 = (u & 0x80000000u) ? (u & 0x7fffffffu) : ~u;
        mv = __uint_as_float(w2);
        if (mv != mv || fabsf(mv) > 3.0e38f) mv = 0.f;
      }
      float ex = __expf(sh_e[it] - mv);
      sh_e[it] = ex;
      atomicAdd(&sh_d[d * NH + hd], ex);
    }
    __syncthreads();

    // v = h@Wv  (overwrites q buffer; q no longer needed) — reuses hoisted hfr
    for (int i = 0; i < 16; ++i) {
      v8f a0 = {0.f, 0.f, 0.f, 0.f, 0.f, 0.f, 0.f, 0.f};
      a0 = wtile_pre(hfr, 8, Wv, HID, i * 16, 0, a0);
      store_tile_bf16(sh_q, HID, wave * 16, i * 16, a0);
    }
    __syncthreads();

    // agg = segment_sum(alpha*v) in channel halves; C = agg @ Wo accumulated in regs
    v8f acc[16];
#pragma unroll
    for (int i = 0; i < 16; ++i) acc[i] = (v8f){0.f, 0.f, 0.f, 0.f, 0.f, 0.f, 0.f, 0.f};
    float* aggf = sh_kagg;
    const __bf16* vv = sh_q;
    for (int h2 = 0; h2 < 2; ++h2) {
      for (int i = tid; i < NN * 128; i += 256) aggf[i] = 0.f;
      __syncthreads();
      for (int it = tid; it < NEDGE * 4; it += 256) {
        int e = it >> 2, hq = it & 3;
        int hd = h2 * 4 + hq;
        int s = srcp[e] & 127, d = dstp[e] & 127;
        float alpha = sh_e[e * NH + hd] / fmaxf(sh_d[d * NH + hd], 1e-9f);
        for (int j = 0; j < HD; ++j)
          atomicAdd(&aggf[d * 128 + hq * HD + j], alpha * (float)vv[s * HID + hd * HD + j]);
      }
      __syncthreads();
      v16bf gfr[4];
      load_afrags(aggf, 128, wave * 16, gfr, 4);
#pragma unroll
      for (int i = 0; i < 16; ++i)
        acc[i] = wtile_pre(gfr, 4, Wo, HID, i * 16, h2 * 128, acc[i]);
      __syncthreads();
    }
    // h += relu(C)
#pragma unroll
    for (int i = 0; i < 16; ++i)
#pragma unroll
      for (int r = 0; r < 8; ++r)
        sh_h[(wave * 16 + r + 8 * half) * HID + i * 16 + mlan] += fmaxf(acc[i][r], 0.f);
    __syncthreads();
  }

  // ---- edge-type pooling + zcat @ Wc
  float* csum = sh_e;   // 11*256 floats
  float* cnt  = sh_d;   // first 11
  for (int i = tid; i < CCAT * HID; i += 256) csum[i] = 0.f;
  if (tid < CCAT) cnt[tid] = 0.f;
  __syncthreads();
  for (int it = tid; it < NEDGE * HID; it += 256) {
    int e = it >> 8, c = it & 255;
    int s = srcp[e] & 127;
    int ty = ET[(size_t)g * NEDGE + e];
    ty = ty < 0 ? 0 : (ty > CCAT - 1 ? CCAT - 1 : ty);
    atomicAdd(&csum[ty * HID + c], sh_h[s * HID + c]);
  }
  for (int e = tid; e < NEDGE; e += 256) {
    int ty = ET[(size_t)g * NEDGE + e];
    ty = ty < 0 ? 0 : (ty > CCAT - 1 ? CCAT - 1 : ty);
    atomicAdd(&cnt[ty], 1.f);
  }
  __syncthreads();
  {
    int j = tid;
    float o = bc[j];
    for (int i = 0; i < HID; ++i) o += sanit(sh_h[i]) * Wc[(size_t)i * HID + j];
    for (int c = 0; c < CCAT; ++c) {
      float ic = 1.f / fmaxf(cnt[c], 1.f);
      for (int i = 0; i < HID; ++i)
        o += sanit(csum[c * HID + i] * ic) * Wc[(size_t)(HID + c * HID + i) * HID + j];
    }
    Z[(size_t)g * HID + j] = o;
  }
}

// ---------------------------------------------------------------- assemble temporal z: [cls ; Z] per batch
__global__ void k_assemble(const float* Z, const float* cls, float* z, unsigned short* zb_) {
  int r = blockIdx.x, c = threadIdx.x;
  float v = 0.f;
  if (r < MROWS) {
    int b = r / SSEQ, s = r % SSEQ;
    v = (s == 0) ? cls[c] : Z[(size_t)(b * 64 + (s - 1)) * HID + c];
  }
  z[(size_t)r * HID + c] = v;
  ((__bf16*)zb_)[(size_t)r * HID + c] = (__bf16)v;
}

// ---------------------------------------------------------------- generic bf16 WMMA GEMM (wave = 16x16 tile)
__global__ __launch_bounds__(256)
void k_gemm(const unsigned short* A_, int lda, const unsigned short* Bt_, int ldb,
            const float* bias, float* Cf, unsigned short* Cb_,
            int Mvalid, int Mtiles, int Ntiles, int K, int relu) {
  const __bf16* A  = (const __bf16*)A_;
  const __bf16* Bt = (const __bf16*)Bt_;
  int wave = threadIdx.x >> 5;
  int lane = threadIdx.x & 31;
  int t = blockIdx.x * 8 + wave;
  if (t >= Mtiles * Ntiles) return;
  int tm = t / Ntiles, tn = t % Ntiles;
  int half = lane >> 4, m = lane & 15;
  int N = Ntiles * 16;
  v8f acc = {0.f, 0.f, 0.f, 0.f, 0.f, 0.f, 0.f, 0.f};
  for (int k0 = 0; k0 < K; k0 += 32) {
    const __bf16* pa = A + (size_t)(tm * 16 + m) * lda + k0 + 16 * half;
    v8bf lo = *(const v8bf*)pa;
    v8bf hi = *(const v8bf*)(pa + 8);
    v16bf a = __builtin_shufflevector(lo, hi, 0, 1, 2, 3, 4, 5, 6, 7, 8, 9, 10, 11, 12, 13, 14, 15);
    const __bf16* pb = Bt + (size_t)(tn * 16 + m) * ldb + k0 + 16 * half;
    v16bf b = *(const v16bf*)pb;
    acc = __builtin_amdgcn_wmma_f32_16x16x32_bf16(false, a, false, b, (short)0, acc, false, false);
  }
#pragma unroll
  for (int r = 0; r < 8; ++r) {
    int row = tm * 16 + r + 8 * half;
    int col = tn * 16 + m;
    if (row < Mvalid) {
      float v = acc[r] + (bias ? bias[col] : 0.f);
      if (relu) v = fmaxf(v, 0.f);
      if (Cf) Cf[(size_t)row * N + col] = v;
      if (Cb_) ((__bf16*)Cb_)[(size_t)row * N + col] = (__bf16)v;
    }
  }
}

// ---------------------------------------------------------------- temporal attention, 1 block per (b, head)
__global__ __launch_bounds__(128)
void k_attn(const float* qkv, unsigned short* ob_) {
  __bf16* ob = (__bf16*)ob_;
  int b = blockIdx.x >> 3, hd = blockIdx.x & 7;
  __shared__ float kk[SSEQ * HD], vvv[SSEQ * HD];
  for (int i = threadIdx.x; i < SSEQ * HD; i += 128) {
    int s = i >> 5, d2 = i & 31;
    size_t row = (size_t)(b * SSEQ + s) * (3 * HID);
    kk[i]  = qkv[row + HID + hd * HD + d2];
    vvv[i] = qkv[row + 2 * HID + hd * HD + d2];
  }
  __syncthreads();
  int s = threadIdx.x;
  if (s < SSEQ) {
    float q[HD];
    size_t row = (size_t)(b * SSEQ + s) * (3 * HID) + hd * HD;
    for (int d2 = 0; d2 < HD; ++d2) q[d2] = qkv[row + d2];
    float lg[SSEQ];
    float mx = -3.4e38f;
    for (int t2 = 0; t2 < SSEQ; ++t2) {
      float dot = 0.f;
      for (int d2 = 0; d2 < HD; ++d2) dot += q[d2] * kk[t2 * HD + d2];
      dot *= 0.17677669529663687f;
      lg[t2] = dot;
      mx = fmaxf(mx, dot);
    }
    float sum = 0.f;
    for (int t2 = 0; t2 < SSEQ; ++t2) { float e2 = __expf(lg[t2] - mx); lg[t2] = e2; sum += e2; }
    float inv = 1.f / sum;
    size_t orow = (size_t)(b * SSEQ + s) * HID + hd * HD;
    for (int d2 = 0; d2 < HD; ++d2) {
      float o = 0.f;
      for (int t2 = 0; t2 < SSEQ; ++t2) o += lg[t2] * vvv[t2 * HD + d2];
      ob[orow + d2] = (__bf16)(o * inv);
    }
  }
}

// ---------------------------------------------------------------- z = LN(z + tmp)*g + b ; 1 block per row
__global__ __launch_bounds__(256)
void k_ln(float* z, const float* tmp, const float* gg, const float* bb, unsigned short* zb_) {
  int r = blockIdx.x, c = threadIdx.x;
  __shared__ float red[256];
  float v = z[(size_t)r * HID + c] + tmp[(size_t)r * HID + c];
  red[c] = v;
  __syncthreads();
  for (int off = 128; off > 0; off >>= 1) { if (c < off) red[c] += red[c + off]; __syncthreads(); }
  float mean = red[0] / (float)HID;
  __syncthreads();
  float dv = v - mean;
  red[c] = dv * dv;
  __syncthreads();
  for (int off = 128; off > 0; off >>= 1) { if (c < off) red[c] += red[c + off]; __syncthreads(); }
  float var = red[0] / (float)HID;
  float o = dv * rsqrtf(var + 1e-5f) * gg[c] + bb[c];
  z[(size_t)r * HID + c] = o;
  ((__bf16*)zb_)[(size_t)r * HID + c] = (__bf16)o;
}

// ---------------------------------------------------------------- out = h_cls @ Wp + bp
__global__ void k_final(const float* z, const float* Wp, const float* bp, float* out) {
  int b = blockIdx.x, j = threadIdx.x;
  float acc = bp[j];
  const float* zr = z + (size_t)(b * SSEQ) * HID;
  for (int k = 0; k < HID; ++k) acc += zr[k] * Wp[(size_t)k * HID + j];
  out[(size_t)b * HID + j] = acc;
}

// ---------------------------------------------------------------- host
extern "C" void kernel_launch(void* const* d_in, const int* in_sizes, int n_in,
                              void* d_out, int out_size, void* d_ws, size_t ws_size,
                              hipStream_t stream) {
  (void)in_sizes; (void)n_in; (void)out_size; (void)ws_size;
  const float* X   = (const float*)d_in[0];
  const int*   EI  = (const int*)d_in[1];
  const int*   ET  = (const int*)d_in[2];
  const float* GEO = (const float*)d_in[3];
  const float* WIN = (const float*)d_in[4];
  const float* BIN = (const float*)d_in[5];
  const float* WQ  = (const float*)d_in[6];
  const float* WK  = (const float*)d_in[7];
  const float* WV  = (const float*)d_in[8];
  const float* WO  = (const float*)d_in[9];
  const float* WG  = (const float*)d_in[10];
  const float* WC  = (const float*)d_in[11];
  const float* BC  = (const float*)d_in[12];
  const float* CLS = (const float*)d_in[13];
  const float* WQKV = (const float*)d_in[14];
  const float* BQKV = (const float*)d_in[15];
  const float* WOT  = (const float*)d_in[16];
  const float* BOT  = (const float*)d_in[17];
  const float* G1   = (const float*)d_in[18];
  const float* B1   = (const float*)d_in[19];
  const float* W1   = (const float*)d_in[20];
  const float* BF1  = (const float*)d_in[21];
  const float* W2   = (const float*)d_in[22];
  const float* BF2  = (const float*)d_in[23];
  const float* G2   = (const float*)d_in[24];
  const float* B2   = (const float*)d_in[25];
  const float* WP   = (const float*)d_in[26];
  const float* BP   = (const float*)d_in[27];

  char* wp = (char*)d_ws;
  auto alloc = [&](size_t bytes) { char* p = wp; wp += (bytes + 255) & ~(size_t)255; return p; };
  unsigned short* WqT   = (unsigned short*)alloc((size_t)3 * HID * HID * 2);
  unsigned short* WkT   = (unsigned short*)alloc((size_t)3 * HID * HID * 2);
  unsigned short* WvT   = (unsigned short*)alloc((size_t)3 * HID * HID * 2);
  unsigned short* WoT   = (unsigned short*)alloc((size_t)3 * HID * HID * 2);
  unsigned short* WqkvT = (unsigned short*)alloc((size_t)2 * 3 * HID * HID * 2);
  unsigned short* WotT  = (unsigned short*)alloc((size_t)2 * HID * HID * 2);
  unsigned short* W1T   = (unsigned short*)alloc((size_t)2 * HID * DFF * 2);
  unsigned short* W2T   = (unsigned short*)alloc((size_t)2 * DFF * HID * 2);
  float*          Zbuf  = (float*)alloc((size_t)NG * HID * 4);
  float*          zf    = (float*)alloc((size_t)MPAD * HID * 4);
  unsigned short* zb    = (unsigned short*)alloc((size_t)MPAD * HID * 2);
  float*          qkvf  = (float*)alloc((size_t)MPAD * 3 * HID * 4);
  unsigned short* ob    = (unsigned short*)alloc((size_t)MPAD * HID * 2);
  unsigned short* ffb   = (unsigned short*)alloc((size_t)MPAD * DFF * 2);
  float*          tmpf  = (float*)alloc((size_t)MPAD * HID * 4);

  auto tr = [&](const float* W, unsigned short* Wt, int K, int N) {
    int tot = K * N;
    k_w2bf_t<<<(tot + 255) / 256, 256, 0, stream>>>(W, Wt, K, N);
  };
  for (int l = 0; l < 3; ++l) {
    tr(WQ + (size_t)l * HID * HID, WqT + (size_t)l * HID * HID, HID, HID);
    tr(WK + (size_t)l * HID * HID, WkT + (size_t)l * HID * HID, HID, HID);
    tr(WV + (size_t)l * HID * HID, WvT + (size_t)l * HID * HID, HID, HID);
    tr(WO + (size_t)l * HID * HID, WoT + (size_t)l * HID * HID, HID, HID);
  }
  for (int l = 0; l < 2; ++l) {
    tr(WQKV + (size_t)l * HID * 3 * HID, WqkvT + (size_t)l * 3 * HID * HID, HID, 3 * HID);
    tr(WOT + (size_t)l * HID * HID, WotT + (size_t)l * HID * HID, HID, HID);
    tr(W1 + (size_t)l * HID * DFF, W1T + (size_t)l * HID * DFF, HID, DFF);
    tr(W2 + (size_t)l * DFF * HID, W2T + (size_t)l * DFF * HID, DFF, HID);
  }

  // fused spatial GNN: 512 graphs, 1 WG each
  k_spatial<<<NG, 256, 0, stream>>>(X, EI, ET, GEO, WIN, BIN, WqT, WkT, WvT, WoT, WG, WC, BC, Zbuf);

  // temporal transformer
  k_assemble<<<MPAD, 256, 0, stream>>>(Zbuf, CLS, zf, zb);
  const int Mt = MPAD / 16;  // 33
  for (int l = 0; l < 2; ++l) {
    // qkv = z @ Wqkv + bqkv   (M=520 N=768 K=256)
    k_gemm<<<(Mt * 48 + 7) / 8, 256, 0, stream>>>(zb, HID, WqkvT + (size_t)l * 3 * HID * HID, HID,
                                                  BQKV + (size_t)l * 3 * HID, qkvf, nullptr,
                                                  MROWS, Mt, 48, HID, 0);
    k_attn<<<NBATCH * NH, 128, 0, stream>>>(qkvf, ob);
    // tmp = o @ Wot + bot
    k_gemm<<<(Mt * 16 + 7) / 8, 256, 0, stream>>>(ob, HID, WotT + (size_t)l * HID * HID, HID,
                                                  BOT + (size_t)l * HID, tmpf, nullptr,
                                                  MROWS, Mt, 16, HID, 0);
    k_ln<<<MROWS, 256, 0, stream>>>(zf, tmpf, G1 + (size_t)l * HID, B1 + (size_t)l * HID, zb);
    // ff1 = relu(z @ W1 + bf1) -> bf16
    k_gemm<<<(Mt * 128 + 7) / 8, 256, 0, stream>>>(zb, HID, W1T + (size_t)l * HID * DFF, HID,
                                                   BF1 + (size_t)l * DFF, nullptr, ffb,
                                                   MROWS, Mt, 128, HID, 1);
    // tmp = ff1 @ W2 + bf2
    k_gemm<<<(Mt * 16 + 7) / 8, 256, 0, stream>>>(ffb, DFF, W2T + (size_t)l * DFF * HID, DFF,
                                                  BF2 + (size_t)l * HID, tmpf, nullptr,
                                                  MROWS, Mt, 16, DFF, 0);
    k_ln<<<MROWS, 256, 0, stream>>>(zf, tmpf, G2 + (size_t)l * HID, B2 + (size_t)l * HID, zb);
  }
  k_final<<<NBATCH, 256, 0, stream>>>(zf, WP, BP, (float*)d_out);
}
